// NNConvModel_14319420964875
// MI455X (gfx1250) — matrix operand order repaired and checked by reading, not compiled, not run on hardware
//
#include <hip/hip_runtime.h>
#include <hip/hip_bf16.h>

typedef __attribute__((ext_vector_type(16))) __bf16 v16bf;
typedef __attribute__((ext_vector_type(8)))  __bf16 v8bf;
typedef __attribute__((ext_vector_type(8)))  float  v8f;

constexpr int NN     = 100000;   // nodes
constexpr int EE     = 300000;   // edges
constexpr int EE_PAD = 300032;   // padded to 64-row GEMM tile
constexpr int BB     = 4096;     // graphs
constexpr int GED    = 128;

__device__ __forceinline__ float sigmf(float x) { return 1.0f / (1.0f + __expf(-x)); }

// ---------------- utility ----------------
__global__ void k_zero(float* p, int n) {
  int i = blockIdx.x * blockDim.x + threadIdx.x;
  if (i < n) p[i] = 0.0f;
}

// ---------------- lin0 + ReLU : nf = relu(x @ lin0_w^T + b) ----------------
__global__ void k_lin0(const float* __restrict__ x, const float* __restrict__ w,
                       const float* __restrict__ b, float* __restrict__ nf) {
  int idx = blockIdx.x * blockDim.x + threadIdx.x;
  if (idx >= NN * 32) return;
  int n = idx >> 5, i = idx & 31;
  float acc = b[i];
  const float* xr = x + n * 16;
  const float* wr = w + i * 16;
  #pragma unroll
  for (int k = 0; k < 16; ++k) acc += xr[k] * wr[k];
  nf[idx] = fmaxf(acc, 0.0f);
}

// ---------------- edge MLP layer1 -> bf16 ehid (E x 128) ----------------
__global__ void k_ehid(const float* __restrict__ ea, const float* __restrict__ w1,
                       const float* __restrict__ b1, __bf16* __restrict__ ehid) {
  long long idx = (long long)blockIdx.x * blockDim.x + threadIdx.x;
  if (idx >= (long long)EE * 128) return;
  int e = (int)(idx >> 7), j = (int)(idx & 127);
  const float* a = ea + e * 3;
  float acc = b1[j] + a[0] * w1[j*3+0] + a[1] * w1[j*3+1] + a[2] * w1[j*3+2];
  ehid[idx] = (__bf16)fmaxf(acc, 0.0f);
}

// ---------------- enn_w2 (1024x128 f32) -> bf16 copy (row j = K-contiguous) ----------------
__global__ void k_w2bf(const float* __restrict__ w2, __bf16* __restrict__ Bm) {
  int idx = blockIdx.x * blockDim.x + threadIdx.x;
  if (idx < 1024 * 128) Bm[idx] = (__bf16)w2[idx];
}

// ---------------- degree / inverse degree ----------------
__global__ void k_deg(const int* __restrict__ ei, float* __restrict__ deg) {
  int e = blockIdx.x * blockDim.x + threadIdx.x;
  if (e < EE) atomicAdd(&deg[ei[EE + e]], 1.0f);
}
__global__ void k_invdeg(float* deg) {
  int n = blockIdx.x * blockDim.x + threadIdx.x;
  if (n < NN) { float d = deg[n]; deg[n] = d > 0.0f ? 1.0f / d : 0.0f; }
}

// ---------------- segment starts from sorted batch ----------------
__global__ void k_seg(const int* __restrict__ batch, int* __restrict__ seg) {
  int n = blockIdx.x * blockDim.x + threadIdx.x;
  if (n >= NN) return;
  int b = batch[n];
  if (n == 0) { seg[b] = 0; seg[BB] = NN; }
  else if (batch[n - 1] != b) seg[b] = n;
}

// ---------------- WMMA GEMM: W_e = ehid(E_pad x128) @ enn_w2^T(128x1024) + b2, bf16 out ----
// A staged in LDS (64x128 bf16); B fragments read straight from global Bm
// (Bm[j][k] = enn_w2 row j, so each lane's 16 B elements are K-contiguous).
// Edge dim padded to EE_PAD: no guards anywhere.
__global__ void __launch_bounds__(256) k_enn_gemm(
    const __bf16* __restrict__ ehid, const __bf16* __restrict__ Bm,
    const float* __restrict__ b2, __bf16* __restrict__ We) {
  __shared__ __bf16 As[64 * 128];
  int tid = threadIdx.x;
  int eBase = blockIdx.y * 64;
  int cBase = blockIdx.x * 32;

  unsigned int* AsU = (unsigned int*)As;                 // 4096 dwords
  const unsigned int* eU = (const unsigned int*)ehid + (size_t)eBase * 64;
  #pragma unroll
  for (int it = 0; it < 16; ++it) {
    int idx = tid + it * 256;
    AsU[idx] = eU[idx];                                  // tile is row-contiguous
  }
  __syncthreads();

  int lane = tid & 31, wid = tid >> 5;
  int r0 = (wid & 3) * 16, c0 = (wid >> 2) * 16;
  int mrow = r0 + (lane & 15);
  int hA = (lane >> 4) * 8;          // A K-offset for upper lane half
  int hB = (lane >> 4) * 8;          // B K-offset for upper lane half
  int colB = cBase + c0 + (lane & 15);
  const __bf16* Arow = As + mrow * 128;
  const __bf16* Brow = Bm + (size_t)colB * 128;

  v8f c = {};
  #pragma unroll
  for (int kb = 0; kb < 128; kb += 32) {
    const v8bf* Ap = (const v8bf*)(Arow + kb + hA);   // 16B aligned
    const v8bf* Bp = (const v8bf*)(Brow + kb + hB);   // 16B aligned
    v16bf a = __builtin_shufflevector(Ap[0], Ap[2],
        0,1,2,3,4,5,6,7,8,9,10,11,12,13,14,15);       // K: hA+[0..8) , hA+16+[0..8)
    v16bf b = __builtin_shufflevector(Bp[0], Bp[2],
        0,1,2,3,4,5,6,7,8,9,10,11,12,13,14,15);       // K: hB+[0..8) , hB+16+[0..8)
    c = __builtin_amdgcn_wmma_f32_16x16x32_bf16(false, a, false, b, (short)0, c, false, false);
  }

  int row0 = eBase + r0 + ((lane >> 4) ? 8 : 0);      // ISA f32 C/D layout
  float bias = b2[colB];
  __bf16* Wp = We + (size_t)row0 * 1024 + colB;
  #pragma unroll
  for (int r = 0; r < 8; ++r)
    Wp[r * 1024] = (__bf16)(c[r] + bias);             // unconditional, imm offsets
}

// ---------------- message: one wave per edge, lane = output dim ----------------
__global__ void __launch_bounds__(256) k_msg(const int* __restrict__ ei,
    const float* __restrict__ nf, const __bf16* __restrict__ We,
    float* __restrict__ agg) {
  int lane = threadIdx.x & 31;
  int e = blockIdx.x * 8 + (threadIdx.x >> 5);
  if (e >= EE) return;
  int s = ei[e], d = ei[EE + e];
  const __bf16* W = We + (size_t)e * 1024;
  __builtin_prefetch(W + lane * 32, 0, 0);
  float x = nf[s * 32 + lane];
  float acc = 0.0f;
  #pragma unroll
  for (int i = 0; i < 32; ++i) {
    float xi = __shfl(x, i, 32);
    acc += xi * (float)W[i * 32 + lane];
  }
  atomicAdd(&agg[d * 32 + lane], acc);
}

// ---------------- NNConv epilogue + GRU (out == h invariant) ----------------
__global__ void __launch_bounds__(256) k_gru(
    float* __restrict__ nf, const float* __restrict__ agg,
    const float* __restrict__ invdeg,
    const float* __restrict__ root_w, const float* __restrict__ conv_b,
    const float* __restrict__ wih, const float* __restrict__ whh,
    const float* __restrict__ bih, const float* __restrict__ bhh) {
  __shared__ float s_root[32 * 32], s_wih[96 * 32], s_whh[96 * 32];
  __shared__ float s_cb[32], s_bih[96], s_bhh[96];
  int tid = threadIdx.x;
  for (int i = tid; i < 1024; i += 256) s_root[i] = root_w[i];
  for (int i = tid; i < 3072; i += 256) { s_wih[i] = wih[i]; s_whh[i] = whh[i]; }
  if (tid < 32) s_cb[tid] = conv_b[tid];
  if (tid < 96) { s_bih[tid] = bih[tid]; s_bhh[tid] = bhh[tid]; }
  __syncthreads();
  int n = blockIdx.x * 256 + tid;
  if (n >= NN) return;
  float o[32], m[32];
  float iv = invdeg[n];
  #pragma unroll
  for (int i = 0; i < 32; ++i) o[i] = nf[n * 32 + i];
  #pragma unroll
  for (int i = 0; i < 32; ++i) {
    float acc = agg[n * 32 + i] * iv + s_cb[i];
    #pragma unroll
    for (int j = 0; j < 32; ++j) acc += o[j] * s_root[j * 32 + i];
    m[i] = fmaxf(acc, 0.0f);
  }
  for (int i = 0; i < 32; ++i) {
    float gir = s_bih[i],      ghr = s_bhh[i];
    float giz = s_bih[32 + i], ghz = s_bhh[32 + i];
    float gin = s_bih[64 + i], ghn = s_bhh[64 + i];
    #pragma unroll
    for (int j = 0; j < 32; ++j) {
      float mj = m[j], oj = o[j];
      gir += mj * s_wih[i * 32 + j];
      ghr += oj * s_whh[i * 32 + j];
      giz += mj * s_wih[(32 + i) * 32 + j];
      ghz += oj * s_whh[(32 + i) * 32 + j];
      gin += mj * s_wih[(64 + i) * 32 + j];
      ghn += oj * s_whh[(64 + i) * 32 + j];
    }
    float r  = sigmf(gir + ghr);
    float z  = sigmf(giz + ghz);
    float ng = tanhf(gin + r * ghn);
    nf[n * 32 + i] = (1.0f - z) * ng + z * o[i];
  }
}

// ---------------- Set2Set LSTM step (one block per graph) ----------------
__global__ void __launch_bounds__(128) k_lstm(
    const float* __restrict__ qs, float* __restrict__ hh, float* __restrict__ cc,
    const float* __restrict__ wih, const float* __restrict__ whh,
    const float* __restrict__ bih, const float* __restrict__ bhh) {
  __shared__ float G[128];
  int g = blockIdx.x, j = threadIdx.x;
  float acc = bih[j] + bhh[j];
  const float* q = qs + g * 64;
  const float* h = hh + g * 32;
  #pragma unroll
  for (int k = 0; k < 64; ++k) acc += q[k] * wih[j * 64 + k];
  #pragma unroll
  for (int k = 0; k < 32; ++k) acc += h[k] * whh[j * 32 + k];
  G[j] = acc;
  __syncthreads();
  if (j < 32) {
    float ig = sigmf(G[j]);
    float fg = sigmf(G[32 + j]);
    float gg = tanhf(G[64 + j]);
    float og = sigmf(G[96 + j]);
    float c = fg * cc[g * 32 + j] + ig * gg;
    cc[g * 32 + j] = c;
    hh[g * 32 + j] = og * tanhf(c);
  }
}

// ---------------- Set2Set attention + readout (one wave per graph) ----------------
__global__ void __launch_bounds__(32) k_attn(
    const float* __restrict__ nf, const float* __restrict__ hh,
    const int* __restrict__ seg, float* __restrict__ qs) {
  int g = blockIdx.x, lane = threadIdx.x;
  int s = seg[g], t = seg[g + 1];
  float q = hh[g * 32 + lane];
  float mx = -1e30f;
  for (int n = s; n < t; ++n) {
    float v = nf[n * 32 + lane] * q;
    #pragma unroll
    for (int off = 16; off; off >>= 1) v += __shfl_xor(v, off, 32);
    mx = fmaxf(mx, v);
  }
  float denom = 0.0f, r = 0.0f;
  for (int n = s; n < t; ++n) {
    float v = nf[n * 32 + lane] * q;
    #pragma unroll
    for (int off = 16; off; off >>= 1) v += __shfl_xor(v, off, 32);
    float w = __expf(v - mx);
    denom += w;
    r += w * nf[n * 32 + lane];
  }
  r /= denom;
  qs[g * 64 + lane] = q;
  qs[g * 64 + 32 + lane] = r;
}

// ---------------- final: graph_embed = qs @ lin1^T + b; pred = relu(ge) @ lin2^T + b ----
__global__ void __launch_bounds__(128) k_final(
    const float* __restrict__ qs, const float* __restrict__ w1,
    const float* __restrict__ b1, const float* __restrict__ w2,
    const float* __restrict__ b2, float* __restrict__ outp) {
  __shared__ float red[128];
  int g = blockIdx.x, j = threadIdx.x;
  float acc = b1[j];
  const float* q = qs + g * 64;
  #pragma unroll
  for (int k = 0; k < 64; ++k) acc += q[k] * w1[j * 64 + k];
  outp[BB + g * GED + j] = acc;                 // graph_embed
  red[j] = fmaxf(acc, 0.0f) * w2[j];
  __syncthreads();
  for (int off = 64; off; off >>= 1) {
    if (j < off) red[j] += red[j + off];
    __syncthreads();
  }
  if (j == 0) outp[g] = red[0] + b2[0];         // pred
}

extern "C" void kernel_launch(void* const* d_in, const int* in_sizes, int n_in,
                              void* d_out, int out_size, void* d_ws, size_t ws_size,
                              hipStream_t stream) {
  const float* x         = (const float*)d_in[0];
  const float* edge_attr = (const float*)d_in[1];
  const float* lin0_w    = (const float*)d_in[2];
  const float* lin0_b    = (const float*)d_in[3];
  const float* enn_w1    = (const float*)d_in[4];
  const float* enn_b1    = (const float*)d_in[5];
  const float* enn_w2    = (const float*)d_in[6];
  const float* enn_b2    = (const float*)d_in[7];
  const float* root_w    = (const float*)d_in[8];
  const float* conv_b    = (const float*)d_in[9];
  const float* gru_wih   = (const float*)d_in[10];
  const float* gru_whh   = (const float*)d_in[11];
  const float* gru_bih   = (const float*)d_in[12];
  const float* gru_bhh   = (const float*)d_in[13];
  const float* lstm_wih  = (const float*)d_in[14];
  const float* lstm_whh  = (const float*)d_in[15];
  const float* lstm_bih  = (const float*)d_in[16];
  const float* lstm_bhh  = (const float*)d_in[17];
  const float* lin1_w    = (const float*)d_in[18];
  const float* lin1_b    = (const float*)d_in[19];
  const float* lin2_w    = (const float*)d_in[20];
  const float* lin2_b    = (const float*)d_in[21];
  const int*   edge_index= (const int*)d_in[22];
  const int*   batch     = (const int*)d_in[23];
  float* outp = (float*)d_out;
  (void)in_sizes; (void)n_in; (void)out_size; (void)ws_size;

  char* w = (char*)d_ws;
  auto carve = [&](size_t bytes) { char* p = w; w += (bytes + 255) & ~(size_t)255; return p; };
  float*  nf     = (float*)carve(sizeof(float) * (size_t)NN * 32);
  float*  agg    = (float*)carve(sizeof(float) * (size_t)NN * 32);
  float*  invdeg = (float*)carve(sizeof(float) * (size_t)NN);
  float*  qs     = (float*)carve(sizeof(float) * (size_t)BB * 64);
  float*  hh     = (float*)carve(sizeof(float) * (size_t)BB * 32);
  float*  cc     = (float*)carve(sizeof(float) * (size_t)BB * 32);
  int*    seg    = (int*)  carve(sizeof(int) * (BB + 1));
  __bf16* Bm     = (__bf16*)carve(sizeof(__bf16) * 1024 * 128);
  __bf16* eh     = (__bf16*)carve(sizeof(__bf16) * (size_t)EE_PAD * 128);
  __bf16* We     = (__bf16*)carve(sizeof(__bf16) * (size_t)EE_PAD * 1024);

  auto blk = [](long long n, int t) { return (unsigned)((n + t - 1) / t); };

  // init state (must happen every call: harness does not re-poison)
  k_zero<<<blk(NN, 256), 256, 0, stream>>>(invdeg, NN);            // used as deg first
  k_zero<<<blk(BB * 64, 256), 256, 0, stream>>>(qs, BB * 64);
  k_zero<<<blk(BB * 32, 256), 256, 0, stream>>>(hh, BB * 32);
  k_zero<<<blk(BB * 32, 256), 256, 0, stream>>>(cc, BB * 32);
  // zero the padded ehid tail (32 rows * 128 bf16 = 2048 dwords)
  k_zero<<<blk((EE_PAD - EE) * 128 / 2, 256), 256, 0, stream>>>(
      (float*)(eh + (size_t)EE * 128), (EE_PAD - EE) * 128 / 2);

  k_lin0<<<blk((long long)NN * 32, 256), 256, 0, stream>>>(x, lin0_w, lin0_b, nf);
  k_ehid<<<blk((long long)EE * 128, 256), 256, 0, stream>>>(edge_attr, enn_w1, enn_b1, eh);
  k_w2bf<<<blk(1024 * 128, 256), 256, 0, stream>>>(enn_w2, Bm);
  k_deg<<<blk(EE, 256), 256, 0, stream>>>(edge_index, invdeg);
  k_invdeg<<<blk(NN, 256), 256, 0, stream>>>(invdeg);
  k_seg<<<blk(NN, 256), 256, 0, stream>>>(batch, seg);

  dim3 ggrid(32, EE_PAD / 64);
  k_enn_gemm<<<ggrid, 256, 0, stream>>>(eh, Bm, enn_b2, We);

  for (int it = 0; it < 3; ++it) {
    k_zero<<<blk((long long)NN * 32, 256), 256, 0, stream>>>(agg, NN * 32);
    k_msg<<<blk(EE, 8), 256, 0, stream>>>(edge_index, nf, We, agg);
    k_gru<<<blk(NN, 256), 256, 0, stream>>>(nf, agg, invdeg, root_w, conv_b,
                                            gru_wih, gru_whh, gru_bih, gru_bhh);
  }
  for (int st = 0; st < 3; ++st) {
    k_lstm<<<BB, 128, 0, stream>>>(qs, hh, cc, lstm_wih, lstm_whh, lstm_bih, lstm_bhh);
    k_attn<<<BB, 32, 0, stream>>>(nf, hh, seg, qs);
  }
  k_final<<<BB, 128, 0, stream>>>(qs, lin1_w, lin1_b, lin2_w, lin2_b, outp);
}
